// SwinTransformerBlock_41875931136533
// MI455X (gfx1250) — matrix operand order repaired and checked by reading, not compiled, not run on hardware
//
#include <hip/hip_runtime.h>

// ---------------------------------------------------------------------------
// Swin block on gfx1250: f16 WMMA pipeline.
//   ln1+shift+window -> qkv GEMM -> window attention (2x WMMA + softmax)
//   -> proj GEMM -> ln2+residual -> fc1 GEMM (GELU) -> fc2 GEMM -> residual
// ---------------------------------------------------------------------------

typedef __attribute__((ext_vector_type(16))) _Float16 v16h;
typedef __attribute__((ext_vector_type(8)))  _Float16 v8h;
typedef __attribute__((ext_vector_type(8)))  float    v8f;

#define WSZ     7
#define SHIFT_  3
#define HEADS_  3
#define C_      96
#define HID_    384
#define IMG_    224
#define HW_     (IMG_*IMG_)
#define NTOK_   49
#define NWIN_   (8*32*32)        /* 8192 windows */
#define M_TOK_  (NWIN_*NTOK_)    /* 401408 tokens, window-ordered */

__device__ __forceinline__ v16h vzero16() {
  v16h r;
#pragma unroll
  for (int i = 0; i < 16; ++i) r[i] = (_Float16)0.0f;
  return r;
}
__device__ __forceinline__ v8f vzero8f() {
  v8f r;
#pragma unroll
  for (int i = 0; i < 8; ++i) r[i] = 0.0f;
  return r;
}
__device__ __forceinline__ v16h vcat(v8h lo, v8h hi) {
  return __builtin_shufflevector(lo, hi, 0,1,2,3,4,5,6,7,8,9,10,11,12,13,14,15);
}

// A fragment (16x32 f16, row-major source): lane L -> row (L&15);
// halves 0..15 = K (L>>4)*16 .. +15 (contiguous).
__device__ __forceinline__ v16h ldA(const _Float16* p, int ld, int lane) {
  const _Float16* q = p + (size_t)(lane & 15) * ld + ((lane >> 4) & 1) * 16;
  return vcat(*(const v8h*)q, *(const v8h*)(q + 8));
}
__device__ __forceinline__ v16h ldA_g(const _Float16* p, int ld, int lane,
                                      int row0, int nrows) {
  int r = row0 + (lane & 15);
  if (r >= nrows) return vzero16();
  const _Float16* q = p + (size_t)r * ld + ((lane >> 4) & 1) * 16;
  return vcat(*(const v8h*)q, *(const v8h*)(q + 8));
}
// B fragment (32x16 f16) loaded from (N,K) row-major (i.e. B^T):
// lane L -> col (L&15); halves 0..15 = K (L>>4)*16 .. +15 (contiguous).
__device__ __forceinline__ v16h ldB(const _Float16* p, int ld, int lane) {
  const _Float16* q = p + (size_t)(lane & 15) * ld + ((lane >> 4) & 1) * 16;
  return vcat(*(const v8h*)q, *(const v8h*)(q + 8));
}
__device__ __forceinline__ v16h ldB_g(const _Float16* p, int ld, int lane,
                                      int col0, int ncols) {
  int n = col0 + (lane & 15);
  if (n >= ncols) return vzero16();
  const _Float16* q = p + (size_t)n * ld + ((lane >> 4) & 1) * 16;
  return vcat(*(const v8h*)q, *(const v8h*)(q + 8));
}

#define WMMA(a, b, c) \
  __builtin_amdgcn_wmma_f32_16x16x32_f16(false, (a), false, (b), (short)0, (c), false, false)

// --------------------------- weight prep -----------------------------------
// src is (K,N) f32 row-major; dst is (N,K) f16 row-major (transposed).
__global__ void __launch_bounds__(256) transpose_w(const float* __restrict__ src,
                                                   _Float16* __restrict__ dst,
                                                   int K, int N) {
  int i = blockIdx.x * 256 + threadIdx.x;
  if (i < N * K) {
    int n = i / K, k = i % K;
    dst[(size_t)n * K + k] = (_Float16)src[(size_t)k * N + n];
  }
}

// ------------------ LN1 + cyclic shift + window partition ------------------
__global__ void __launch_bounds__(256) ln1_shift_window(
    const float* __restrict__ x, const float* __restrict__ g,
    const float* __restrict__ b, _Float16* __restrict__ xw) {
  int p  = blockIdx.x * 256 + threadIdx.x;            // (batch, spatial)
  int bi = p / HW_;
  int hw = p % HW_;
  int hh = hw / IMG_, ww = hw % IMG_;
  const float* xp = x + (size_t)bi * C_ * HW_ + hw;
  float s = 0.f, ss = 0.f;
  for (int c = 0; c < C_; ++c) { float v = xp[(size_t)c * HW_]; s += v; ss += v * v; }
  float mu = s * (1.0f / C_);
  float rs = rsqrtf(fmaxf(ss * (1.0f / C_) - mu * mu, 0.0f) + 1e-5f);
  int hs = hh - SHIFT_; if (hs < 0) hs += IMG_;
  int vs = ww - SHIFT_; if (vs < 0) vs += IMG_;
  size_t row = ((size_t)bi * 1024 + (hs / WSZ) * 32 + (vs / WSZ)) * NTOK_
             + (hs % WSZ) * WSZ + (vs % WSZ);
  _Float16* op = xw + row * C_;
  for (int c = 0; c < C_; ++c) {
    float v = xp[(size_t)c * HW_];
    op[c] = (_Float16)((v - mu) * rs * g[c] + b[c]);
  }
}

// ------------------------ generic WMMA GEMM --------------------------------
// D(MxN,f16) = act(A(MxK,f16) @ Bt^T + bias). Bt is (N,K). Block: 4 waves,
// tile 128(M) x 32(N); weight slab staged in LDS with b128 copies.
__global__ void __launch_bounds__(128) gemm_f16(
    const _Float16* __restrict__ A, const _Float16* __restrict__ Bt,
    const float* __restrict__ bias, _Float16* __restrict__ D,
    int M, int N, int K, int act) {
  __shared__ _Float16 sB[32 * 384];                   // 24 KB max slab
  const int tid  = threadIdx.x;
  const int lane = tid & 31;
  const int wv   = tid >> 5;
  const int n0   = blockIdx.y * 32;
  const int m0   = blockIdx.x * 128 + wv * 32;

  { // cooperative slab copy: 32 rows x K halves, contiguous in Bt
    const uint4* src = (const uint4*)(Bt + (size_t)n0 * K);
    uint4* dst = (uint4*)sB;
    const int n128 = (32 * K) >> 3;
    for (int i = tid; i < n128; i += 128) dst[i] = src[i];
  }
  __syncthreads();

  v8f acc00 = vzero8f(), acc01 = vzero8f(), acc10 = vzero8f(), acc11 = vzero8f();
  for (int k0 = 0; k0 < K; k0 += 32) {
    v16h a0 = ldA(A + (size_t)m0 * K + k0, K, lane);
    v16h a1 = ldA(A + (size_t)(m0 + 16) * K + k0, K, lane);
    v16h b0 = ldB(sB + k0, K, lane);
    v16h b1 = ldB(sB + (size_t)16 * K + k0, K, lane);
    acc00 = WMMA(a0, b0, acc00);
    acc01 = WMMA(a0, b1, acc01);
    acc10 = WMMA(a1, b0, acc10);
    acc11 = WMMA(a1, b1, acc11);
  }

  const int rbase = ((lane >> 4) & 1) * 8;
  const int cl    = lane & 15;
#pragma unroll
  for (int i = 0; i < 2; ++i) {
#pragma unroll
    for (int j = 0; j < 2; ++j) {
      v8f acc = (i == 0) ? ((j == 0) ? acc00 : acc01)
                         : ((j == 0) ? acc10 : acc11);
      int col = n0 + j * 16 + cl;
      float bv = bias[col];
#pragma unroll
      for (int r = 0; r < 8; ++r) {
        int row = m0 + i * 16 + rbase + r;
        float v = acc[r] + bv;
        if (act == 1) v = 0.5f * v * (1.0f + erff(v * 0.70710678118f));
        D[(size_t)row * N + col] = (_Float16)v;
      }
    }
  }
}

// ---------------------- windowed attention ---------------------------------
// One block (4 waves) per (window, head): QK^T -> bias/mask/softmax -> PV.
__global__ void __launch_bounds__(128) win_attn(
    const _Float16* __restrict__ qkv, const float* __restrict__ relTable,
    _Float16* __restrict__ out) {
  __shared__ float    Pf[64 * 64];   // logits (f32)
  __shared__ _Float16 Ph[64 * 64];   // softmaxed probs (f16, A-operand layout)
  __shared__ _Float16 Vt[32 * 64];   // V transposed: (dh, token) zero-padded
  const int w    = blockIdx.x;
  const int h    = blockIdx.y;
  const int tid  = threadIdx.x;
  const int lane = tid & 31;
  const int wv   = tid >> 5;
  const _Float16* base = qkv + (size_t)w * NTOK_ * (3 * C_);

  for (int i = tid; i < 32 * 64; i += 128) Vt[i] = (_Float16)0.0f;
  __syncthreads();
  for (int i = tid; i < NTOK_ * 32; i += 128) {
    int t = i >> 5, d = i & 31;
    Vt[d * 64 + t] = base[(size_t)t * (3 * C_) + 2 * C_ + h * 32 + d];
  }

  // ---- Q @ K^T : 16 rows per wave, 64 cols, K=32 (one WMMA per tile) ----
  const int m0 = wv * 16;
  v8f acc[4] = {vzero8f(), vzero8f(), vzero8f(), vzero8f()};
  v16h a = ldA_g(base + h * 32, 3 * C_, lane, m0, NTOK_);
#pragma unroll
  for (int j = 0; j < 4; ++j) {
    v16h b = ldB_g(base + C_ + h * 32, 3 * C_, lane, j * 16, NTOK_);
    acc[j] = WMMA(a, b, acc[j]);
  }

  // ---- scale + relative position bias + pad mask -> LDS ----
  const float scale = 0.17677669529663687f;   // 32^-0.5
  const int rbase = ((lane >> 4) & 1) * 8;
  const int cl    = lane & 15;
#pragma unroll
  for (int j = 0; j < 4; ++j) {
    int col = j * 16 + cl;
#pragma unroll
    for (int r = 0; r < 8; ++r) {
      int row = m0 + rbase + r;
      float v = -1e30f;
      if (row < NTOK_ && col < NTOK_) {
        int qi = row / WSZ, qj = row % WSZ, ki = col / WSZ, kj = col % WSZ;
        int idx = (qi - ki + WSZ - 1) * (2 * WSZ - 1) + (qj - kj + WSZ - 1);
        v = acc[j][r] * scale + relTable[idx * HEADS_ + h];
      }
      Pf[row * 64 + col] = v;
    }
  }
  __syncthreads();

  // ---- per-row softmax (one thread per row) ----
  if (tid < 64) {
    float mx = -1e30f;
    for (int c = 0; c < 64; ++c) mx = fmaxf(mx, Pf[tid * 64 + c]);
    float s = 0.0f;
    for (int c = 0; c < 64; ++c) {
      float e = __expf(Pf[tid * 64 + c] - mx);
      s += e;
      Pf[tid * 64 + c] = e;
    }
    float rs = 1.0f / s;
    for (int c = 0; c < 64; ++c) Ph[tid * 64 + c] = (_Float16)(Pf[tid * 64 + c] * rs);
  }
  __syncthreads();

  // ---- P @ V : K=64 (2 WMMA steps), N=dh=32 (2 tiles) ----
  v8f o0 = vzero8f(), o1 = vzero8f();
#pragma unroll
  for (int kt = 0; kt < 64; kt += 32) {
    v16h a2  = ldA(Ph + (size_t)m0 * 64 + kt, 64, lane);
    v16h vb0 = ldB(Vt + kt, 64, lane);
    v16h vb1 = ldB(Vt + (size_t)16 * 64 + kt, 64, lane);
    o0 = WMMA(a2, vb0, o0);
    o1 = WMMA(a2, vb1, o1);
  }
#pragma unroll
  for (int j = 0; j < 2; ++j) {
    v8f oo = (j == 0) ? o0 : o1;
#pragma unroll
    for (int r = 0; r < 8; ++r) {
      int row = m0 + rbase + r;
      if (row < NTOK_)
        out[((size_t)w * NTOK_ + row) * C_ + h * 32 + j * 16 + cl] = (_Float16)oo[r];
    }
  }
}

// ---------- residual 1 + window-reverse/unshift + LN2 ----------------------
__global__ void __launch_bounds__(256) ln2_res(
    const float* __restrict__ x, const _Float16* __restrict__ projO,
    const float* __restrict__ g, const float* __restrict__ b,
    float* __restrict__ x1, _Float16* __restrict__ xn2) {
  int p  = blockIdx.x * 256 + threadIdx.x;
  int bi = p / HW_;
  int hw = p % HW_;
  int hh = hw / IMG_, ww = hw % IMG_;
  int hs = hh - SHIFT_; if (hs < 0) hs += IMG_;
  int vs = ww - SHIFT_; if (vs < 0) vs += IMG_;
  size_t row = ((size_t)bi * 1024 + (hs / WSZ) * 32 + (vs / WSZ)) * NTOK_
             + (hs % WSZ) * WSZ + (vs % WSZ);
  const float* xp = x + (size_t)bi * C_ * HW_ + hw;
  float* x1p = x1 + (size_t)bi * C_ * HW_ + hw;
  const _Float16* pp = projO + row * C_;
  float s = 0.f, ss = 0.f;
  for (int c = 0; c < C_; ++c) {
    float v = xp[(size_t)c * HW_] + (float)pp[c];
    x1p[(size_t)c * HW_] = v;
    s += v; ss += v * v;
  }
  float mu = s * (1.0f / C_);
  float rs = rsqrtf(fmaxf(ss * (1.0f / C_) - mu * mu, 0.0f) + 1e-5f);
  _Float16* op = xn2 + (size_t)p * C_;
  for (int c = 0; c < C_; ++c) {
    float v = x1p[(size_t)c * HW_];
    op[c] = (_Float16)((v - mu) * rs * g[c] + b[c]);
  }
}

// ----------------------------- residual 2 ----------------------------------
__global__ void __launch_bounds__(256) final_add(
    const float* __restrict__ x1, const _Float16* __restrict__ f2o,
    float* __restrict__ out) {
  size_t i = (size_t)blockIdx.x * 256 + threadIdx.x;
  size_t per = (size_t)C_ * HW_;
  int bi = (int)(i / per);
  size_t rem = i % per;
  int c  = (int)(rem / HW_);
  int hw = (int)(rem % HW_);
  out[i] = x1[i] + (float)f2o[((size_t)bi * HW_ + hw) * C_ + c];
}

// ---------------------------------------------------------------------------
extern "C" void kernel_launch(void* const* d_in, const int* in_sizes, int n_in,
                              void* d_out, int out_size, void* d_ws, size_t ws_size,
                              hipStream_t stream) {
  (void)in_sizes; (void)n_in; (void)out_size; (void)ws_size;
  const float* x        = (const float*)d_in[0];
  const float* n1g      = (const float*)d_in[1];
  const float* n1b      = (const float*)d_in[2];
  const float* qkv_w    = (const float*)d_in[3];
  const float* qkv_b    = (const float*)d_in[4];
  const float* proj_w   = (const float*)d_in[5];
  const float* proj_b   = (const float*)d_in[6];
  const float* relTable = (const float*)d_in[7];
  const float* n2g      = (const float*)d_in[8];
  const float* n2b      = (const float*)d_in[9];
  const float* fc1_w    = (const float*)d_in[10];
  const float* fc1_b    = (const float*)d_in[11];
  const float* fc2_w    = (const float*)d_in[12];
  const float* fc2_b    = (const float*)d_in[13];
  float* out = (float*)d_out;

  char* ws = (char*)d_ws;
  // f16 transposed weights (tiny, < 256 KB total)
  _Float16* wqkv  = (_Float16*)(ws + (0   << 10));
  _Float16* wproj = (_Float16*)(ws + (64  << 10));
  _Float16* wfc1  = (_Float16*)(ws + (128 << 10));
  _Float16* wfc2  = (_Float16*)(ws + (256 << 10));
  // activation buffers (with liveness-based reuse)
  const size_t o_xw   = 1ull << 20;
  const size_t o_qkv  = o_xw   +  77070336ull;   // 401408*96*2
  const size_t o_attn = o_qkv  + 231211008ull;   // 401408*288*2
  const size_t o_proj = o_attn +  77070336ull;
  const size_t o_x1   = o_proj +  77070336ull;
  const size_t o_h    = o_x1   + 154140672ull;   // 401408*96*4
  _Float16* xw    = (_Float16*)(ws + o_xw);
  _Float16* qkvb  = (_Float16*)(ws + o_qkv);
  _Float16* attnO = (_Float16*)(ws + o_attn);
  _Float16* projO = (_Float16*)(ws + o_proj);
  float*    x1    = (float*)   (ws + o_x1);
  _Float16* xn2   = (_Float16*)(ws + o_qkv);     // reuse: qkv dead after attn
  _Float16* hbuf  = (_Float16*)(ws + o_h);
  _Float16* f2o   = (_Float16*)(ws + o_xw);      // reuse: xw dead after qkv GEMM

  // weight prep (f32 -> f16, transpose to (N,K))
  transpose_w<<<dim3((96 * 288 + 255) / 256), 256, 0, stream>>>(qkv_w,  wqkv, 96,  288);
  transpose_w<<<dim3((96 * 96  + 255) / 256), 256, 0, stream>>>(proj_w, wproj, 96,  96);
  transpose_w<<<dim3((96 * 384 + 255) / 256), 256, 0, stream>>>(fc1_w,  wfc1, 96,  384);
  transpose_w<<<dim3((384 * 96 + 255) / 256), 256, 0, stream>>>(fc2_w,  wfc2, 384, 96);

  ln1_shift_window<<<dim3(M_TOK_ / 256), 256, 0, stream>>>(x, n1g, n1b, xw);
  gemm_f16<<<dim3(M_TOK_ / 128, 288 / 32), 128, 0, stream>>>(xw,   wqkv,  qkv_b,  qkvb, M_TOK_, 288, 96,  0);
  win_attn<<<dim3(NWIN_, HEADS_), 128, 0, stream>>>(qkvb, relTable, attnO);
  gemm_f16<<<dim3(M_TOK_ / 128, 96 / 32),  128, 0, stream>>>(attnO, wproj, proj_b, projO, M_TOK_, 96, 96,  0);
  ln2_res<<<dim3(M_TOK_ / 256), 256, 0, stream>>>(x, projO, n2g, n2b, x1, xn2);
  gemm_f16<<<dim3(M_TOK_ / 128, 384 / 32), 128, 0, stream>>>(xn2,  wfc1,  fc1_b,  hbuf, M_TOK_, 384, 96,  1);
  gemm_f16<<<dim3(M_TOK_ / 128, 96 / 32),  128, 0, stream>>>(hbuf, wfc2,  fc2_b,  f2o,  M_TOK_, 96, 384,  0);
  final_add<<<dim3((8 * C_ * HW_) / 256), 256, 0, stream>>>(x1, f2o, out);
}
